// GATMLPModule_63651415326968
// MI455X (gfx1250) — matrix operand order repaired and checked by reading, not compiled, not run on hardware
//
#include <hip/hip_runtime.h>
#include <math.h>

#if __has_builtin(__builtin_amdgcn_global_load_async_to_lds_b128) && \
    __has_builtin(__builtin_amdgcn_s_wait_asynccnt)
#define HAS_ASYNC_LDS 1
#endif

namespace {

constexpr int H_   = 4;
constexpr int Nn   = 15000;   // nodes
constexpr int E1_  = 100000;  // edges graph 1
constexpr int E2_  = 60000;   // edges graph 2
constexpr int NR_  = 13500;   // n_r
constexpr int OUT_ = 64;
constexpr int IN_NODE = 44;   // NCLS + 2
constexpr int IN_EDGE = 2;

typedef __attribute__((ext_vector_type(16))) _Float16 v16h;
typedef __attribute__((ext_vector_type(8)))  float    v8f;

// b128 payload type matching the async-LDS builtin's expected pointee
typedef int b128_t __attribute__((vector_size(16)));
typedef __attribute__((address_space(1))) b128_t* gptr128;
typedef __attribute__((address_space(3))) b128_t* lptr128;

// Convert weight W (KxN f32 row-major) -> Wt (NxK f16 row-major, transposed)
__global__ void cvt_wT(const float* __restrict__ W, _Float16* __restrict__ Wt,
                       int K, int N) {
  long i = (long)blockIdx.x * blockDim.x + threadIdx.x;
  if (i >= (long)N * K) return;
  int n = (int)(i / K), k = (int)(i % K);
  Wt[i] = (_Float16)W[(long)k * N + n];
}

// ---------------------------------------------------------------------------
// GEMM: C = act(A @ W + bias).  A: MxK f32 row-major, Wt: NxK f16 (W^T).
// Block = 256 threads = 8 wave32; block tile 32(M) x 128(N); K step 32.
// Wave (wm,wn) computes rows [wm*16,+16) x cols [wn*32,+32): two WMMAs per
// K-step reusing the A fragment.  B tiles staged via async global->LDS copy
// when available (GLOBAL_LOAD_ASYNC_TO_LDS_B128 + s_wait_asynccnt).
// act: 0 = none, 1 = relu, 2 = leaky-relu(0.01)
// ---------------------------------------------------------------------------
__global__ __launch_bounds__(256)
void gemm_wmma(const float* __restrict__ A, const _Float16* __restrict__ Wt,
               const float* __restrict__ bias, float* __restrict__ C,
               int M, int K, int Ncols, int act)
{
  __shared__ _Float16 As[32][40];    // [row][k]
  __shared__ _Float16 Bs[128][40];   // [col][k]  (row of W^T = col of W)

  const int tid  = threadIdx.x;
  const int lane = tid & 31;
  const int w    = tid >> 5;
  const int wm   = w >> 2;          // 0..1
  const int wn   = w & 3;           // 0..3
  const int m0   = blockIdx.x * 32;
  const int n0   = blockIdx.y * 128;

  v8f acc0 = {}, acc1 = {};
  const int  kTiles = (K + 31) >> 5;
  const bool bFast  = ((K & 31) == 0);

  for (int kt = 0; kt < kTiles; ++kt) {
    const int k0 = kt << 5;

    // ---- stage A tile (32 x 32), f32 -> f16 ----
    if (((K & 3) == 0) && (k0 + 32 <= K)) {
      const int r  = tid >> 3;
      const int c4 = (tid & 7) << 2;
      float4 v = {0.f, 0.f, 0.f, 0.f};
      const int gm = m0 + r;
      if (gm < M) v = *(const float4*)(A + (long)gm * K + k0 + c4);
      As[r][c4 + 0] = (_Float16)v.x;
      As[r][c4 + 1] = (_Float16)v.y;
      As[r][c4 + 2] = (_Float16)v.z;
      As[r][c4 + 3] = (_Float16)v.w;
    } else {
      for (int i = tid; i < 1024; i += 256) {
        int r = i >> 5, k = i & 31;
        int gm = m0 + r, gk = k0 + k;
        As[r][k] = (gm < M && gk < K) ? (_Float16)A[(long)gm * K + gk]
                                      : (_Float16)0.f;
      }
    }

    // ---- stage B tile (128 cols x 32 k), f16 ----
    if (bFast) {
#ifdef HAS_ASYNC_LDS
      for (int s = tid; s < 512; s += 256) {
        const int row = s >> 2, part = s & 3;   // 16B segment per lane
        const int gn  = n0 + row;
        _Float16* ldst = &Bs[row][part * 8];
        if (gn < Ncols) {
          const _Float16* gsrc = Wt + (long)gn * K + k0 + part * 8;
          __builtin_amdgcn_global_load_async_to_lds_b128(
              (gptr128)(unsigned long long)(const char*)gsrc,
              (lptr128)(unsigned int)(unsigned long long)(char*)ldst,
              0, 0);
        } else {
#pragma unroll
          for (int q = 0; q < 8; ++q) ldst[q] = (_Float16)0.f;
        }
      }
      __builtin_amdgcn_s_wait_asynccnt(0);
#else
      for (int s = tid; s < 512; s += 256) {
        const int row = s >> 2, part = s & 3;
        const int gn  = n0 + row;
        _Float16* ldst = &Bs[row][part * 8];
        if (gn < Ncols) {
          const _Float16* gsrc = Wt + (long)gn * K + k0 + part * 8;
#pragma unroll
          for (int q = 0; q < 8; ++q) ldst[q] = gsrc[q];
        } else {
#pragma unroll
          for (int q = 0; q < 8; ++q) ldst[q] = (_Float16)0.f;
        }
      }
#endif
    } else {
      for (int i = tid; i < 4096; i += 256) {
        const int c = i >> 5, k = i & 31;
        const int gn = n0 + c, gk = k0 + k;
        Bs[c][k] = (gn < Ncols && gk < K) ? Wt[(long)gn * K + gk]
                                          : (_Float16)0.f;
      }
    }
    __syncthreads();

    // ---- fragments: contiguous k order (k = 16*half + s), same for A and B ----
    // A 16x32: lanes 0-15 row=lane   K0-15
    //          lanes16-31 row=lane-16 K16-31
    // B 32x16: lanes 0-15 col=lane K0-15; lanes16-31 col=lane-16 K16-31
    const int fr = (wm << 4) + (lane & 15);
    const int hi = lane >> 4;
    const int ka = hi * 16;
    const int kb = hi * 16;
    const int c0 = (wn << 5) + (lane & 15);
    v16h af, bf0, bf1;
#pragma unroll
    for (int h = 0; h < 8; ++h) {
      af[h]     = As[fr][ka + h];
      af[h + 8] = As[fr][ka + 8 + h];
    }
#pragma unroll
    for (int h = 0; h < 16; ++h) bf0[h] = Bs[c0][kb + h];
#pragma unroll
    for (int h = 0; h < 16; ++h) bf1[h] = Bs[c0 + 16][kb + h];

    acc0 = __builtin_amdgcn_wmma_f32_16x16x32_f16(false, af, false, bf0,
                                                  (short)0, acc0, false, false);
    acc1 = __builtin_amdgcn_wmma_f32_16x16x32_f16(false, af, false, bf1,
                                                  (short)0, acc1, false, false);
    __syncthreads();
  }

  // epilogue: C/D layout -> col = lane&15, row = i + 8*(lane>=16)
  const int col = lane & 15;
  const int rb  = (lane >> 4) * 8;
  const int gn0 = n0 + (wn << 5) + col;
  const int gn1 = gn0 + 16;
#pragma unroll
  for (int i = 0; i < 8; ++i) {
    const int gm = m0 + (wm << 4) + rb + i;
    if (gm < M) {
      if (gn0 < Ncols) {
        float v = acc0[i];
        if (bias) v += bias[gn0];
        if (act == 1)      v = fmaxf(v, 0.f);
        else if (act == 2) v = (v > 0.f) ? v : 0.01f * v;
        C[(long)gm * Ncols + gn0] = v;
      }
      if (gn1 < Ncols) {
        float v = acc1[i];
        if (bias) v += bias[gn1];
        if (act == 1)      v = fmaxf(v, 0.f);
        else if (act == 2) v = (v > 0.f) ? v : 0.01f * v;
        C[(long)gm * Ncols + gn1] = v;
      }
    }
  }
}

// --------------------------- elementwise helpers ---------------------------
__global__ void fill_f32(float* __restrict__ p, long n, float v) {
  long i = (long)blockIdx.x * blockDim.x + threadIdx.x;
  if (i < n) p[i] = v;
}
__global__ void copy_f32(const float* __restrict__ s, float* __restrict__ d, long n) {
  long i = (long)blockIdx.x * blockDim.x + threadIdx.x;
  if (i < n) d[i] = s[i];
}
__global__ void relu_ip(float* __restrict__ p, long n) {
  long i = (long)blockIdx.x * blockDim.x + threadIdx.x;
  if (i < n) p[i] = fmaxf(p[i], 0.f);
}
__global__ void add_ip(float* __restrict__ p, const float* __restrict__ q, long n) {
  long i = (long)blockIdx.x * blockDim.x + threadIdx.x;
  if (i < n) p[i] += q[i];
}

// f_out = leaky_relu(f_fij + f_ni[src] + f_nj[dst] + bias), in place over f_fij
__global__ void edge_combine(float* __restrict__ f, const float* __restrict__ fni,
                             const float* __restrict__ fnj, const float* __restrict__ bias,
                             const int* __restrict__ src, const int* __restrict__ dst,
                             int E, int D) {
  long i = (long)blockIdx.x * blockDim.x + threadIdx.x;
  long n = (long)E * D;
  if (i >= n) return;
  int e = (int)(i / D), d = (int)(i % D);
  float v = f[i] + fni[(long)src[e] * D + d] + fnj[(long)dst[e] * D + d] + bias[d];
  f[i] = (v > 0.f) ? v : 0.01f * v;
}

// e[e,h] = sum_d f[e, h*out_e + d] * attn[h, d]
__global__ void attn_logits(const float* __restrict__ f, const float* __restrict__ attn,
                            float* __restrict__ eatt, int E, int out_e) {
  int i = blockIdx.x * blockDim.x + threadIdx.x;
  if (i >= E * H_) return;
  int e = i / H_, h = i % H_;
  const float* fr = f + (long)e * H_ * out_e + (long)h * out_e;
  const float* ar = attn + h * out_e;
  float s = 0.f;
  for (int d = 0; d < out_e; ++d) s += fr[d] * ar[d];
  eatt[i] = s;
}

__device__ inline void atomicMaxF(float* addr, float val) {
  unsigned int* ua = (unsigned int*)addr;
  unsigned int old = *ua, assumed;
  do {
    assumed = old;
    if (__uint_as_float(assumed) >= val) break;
    old = atomicCAS(ua, assumed, __float_as_uint(val));
  } while (old != assumed);
}

__global__ void seg_max(const float* __restrict__ eatt, const int* __restrict__ dst,
                        float* __restrict__ emax, int E) {
  int i = blockIdx.x * blockDim.x + threadIdx.x;
  if (i >= E * H_) return;
  int e = i / H_, h = i % H_;
  atomicMaxF(&emax[dst[e] * H_ + h], eatt[i]);
}

__global__ void fix_max(float* __restrict__ emax, int n) {
  int i = blockIdx.x * blockDim.x + threadIdx.x;
  if (i >= n) return;
  float v = emax[i];
  emax[i] = (v == v && v > -3.0e38f && v < 3.0e38f) ? v : 0.f;
}

__global__ void exp_sum(const float* __restrict__ eatt, const float* __restrict__ emax,
                        const int* __restrict__ dst, float* __restrict__ ex,
                        float* __restrict__ esum, int E) {
  int i = blockIdx.x * blockDim.x + threadIdx.x;
  if (i >= E * H_) return;
  int e = i / H_, h = i % H_;
  float v = expf(eatt[i] - emax[dst[e] * H_ + h]);
  ex[i] = v;
  atomicAdd(&esum[dst[e] * H_ + h], v);
}

__global__ void attn_norm(float* __restrict__ ex, const float* __restrict__ esum,
                          const int* __restrict__ dst, int E) {
  int i = blockIdx.x * blockDim.x + threadIdx.x;
  if (i >= E * H_) return;
  int e = i / H_, h = i % H_;
  float s = esum[dst[e] * H_ + h];
  ex[i] = (s > 0.f) ? ex[i] / s : 0.f;
}

// h_out[dst[e], r] += h_src[src[e], r] * a[e, r/out_n]
__global__ void scatter_h(const float* __restrict__ hsrc, const float* __restrict__ a,
                          const int* __restrict__ src, const int* __restrict__ dst,
                          float* __restrict__ hout, int E, int HD, int out_n) {
  long i = (long)blockIdx.x * blockDim.x + threadIdx.x;
  long n = (long)E * HD;
  if (i >= n) return;
  int e = (int)(i / HD);
  int r = (int)(i % HD);
  int h = r / out_n;
  atomicAdd(&hout[(long)dst[e] * HD + r], hsrc[(long)src[e] * HD + r] * a[e * H_ + h]);
}

__global__ void col_max(const float* __restrict__ nodep, float* __restrict__ feat,
                        int rows, int cols) {
  int c = blockIdx.x * blockDim.x + threadIdx.x;
  if (c >= cols) return;
  float m = -3.4e38f;
  for (int r = 0; r < rows; ++r) m = fmaxf(m, nodep[(long)r * cols + c]);
  feat[c] = m;
}

struct EgatW { const _Float16 *W_fij, *W_ni, *W_nj, *W_node; const float *attn, *bias; };

static inline dim3 g1(long n) { return dim3((unsigned)((n + 255) / 256)); }

} // namespace

extern "C" void kernel_launch(void* const* d_in, const int* in_sizes, int n_in,
                              void* d_out, int out_size, void* d_ws, size_t ws_size,
                              hipStream_t stream) {
  (void)in_sizes; (void)n_in; (void)out_size; (void)ws_size;

  const float* nodeDat = (const float*)d_in[0];
  const float* edgeDat = (const float*)d_in[1];

  // params flattened pytree-style (sorted keys): conv1, conv2, edge_lin, lin1,
  // lin2, lin2a, lin3, node_proj, reg1, reg2, reg3, rq
  int p = 2;
  struct RawE { const float *W_fij, *W_ni, *W_nj, *W_node, *attn, *bias; };
  RawE rconv1, rconv2, rrq;
  auto grabE = [&](RawE& g) {
    g.W_fij  = (const float*)d_in[p++];
    g.W_ni   = (const float*)d_in[p++];
    g.W_nj   = (const float*)d_in[p++];
    g.W_node = (const float*)d_in[p++];
    g.attn   = (const float*)d_in[p++];
    g.bias   = (const float*)d_in[p++];
  };
  grabE(rconv1);
  grabE(rconv2);
  const float* edge_lin_W = (const float*)d_in[p++];
  const float* edge_lin_b = (const float*)d_in[p++];
  const float* lin1_W = (const float*)d_in[p++];
  const float* lin1_b = (const float*)d_in[p++];
  const float* lin2_W = (const float*)d_in[p++];
  const float* lin2_b = (const float*)d_in[p++];
  const float* lin2a_W = (const float*)d_in[p++];
  const float* lin2a_b = (const float*)d_in[p++];
  const float* lin3_W = (const float*)d_in[p++];
  const float* lin3_b = (const float*)d_in[p++];
  const float* nproj_W = (const float*)d_in[p++];
  const float* nproj_b = (const float*)d_in[p++];
  const float* reg1_W = (const float*)d_in[p++];
  const float* reg1_b = (const float*)d_in[p++];
  const float* reg2_W = (const float*)d_in[p++];
  const float* reg2_b = (const float*)d_in[p++];
  const float* reg3_W = (const float*)d_in[p++];
  const float* reg3_b = (const float*)d_in[p++];
  grabE(rrq);
  const int* src1 = (const int*)d_in[p++];
  const int* dst1 = (const int*)d_in[p++];
  const int* src2 = (const int*)d_in[p++];
  const int* dst2 = (const int*)d_in[p++];

  // ---------------- workspace layout (fp32 region) ----------------
  float* ws = (float*)d_ws;
  size_t o = 0;
  auto alloc = [&](long n) { float* r = ws + o; o += (size_t)n; return r; };
  float* node  = alloc((long)Nn * OUT_);
  float* nodet = alloc((long)Nn * OUT_);
  float* edge  = alloc((long)E1_ * OUT_);
  float* elink = alloc((long)E2_ * OUT_);
  float* fni   = alloc((long)Nn * 512);
  float* fnj   = alloc((long)Nn * 512);
  float* ftmp  = alloc((long)E1_ * 512);   // f_fij / f_out for conv1 & rq
  float* fout2 = alloc((long)E1_ * 256);   // conv2 f_out
  float* hsrc  = alloc((long)Nn * 1024);
  float* hA    = alloc((long)Nn * 1024);
  float* hB    = alloc((long)Nn * 1024);
  float* eatt  = alloc((long)E1_ * H_);
  float* exb   = alloc((long)E1_ * H_);
  float* emax  = alloc((long)Nn * H_);
  float* esum  = alloc((long)Nn * H_);
  float* feat  = alloc(64);
  float* x1    = alloc(64);
  float* x2    = alloc(64);
  float* loc   = alloc(64);

  // ---------------- f16 transposed-weight region ----------------
  _Float16* hws = (_Float16*)(ws + o);
  size_t ho = 0;
  auto halloc = [&](long n) { _Float16* r = hws + ho; ho += (size_t)n; return r; };
  _Float16* c1_ni  = halloc(64 * 512);
  _Float16* c1_nj  = halloc(64 * 512);
  _Float16* c1_fij = halloc(64 * 512);
  _Float16* c1_nd  = halloc(64 * 512);
  _Float16* c2_ni  = halloc(512 * 256);
  _Float16* c2_nj  = halloc(512 * 256);
  _Float16* c2_fij = halloc(512 * 256);
  _Float16* c2_nd  = halloc(512 * 1024);
  _Float16* rq_ni  = halloc(64 * 512);
  _Float16* rq_nj  = halloc(64 * 512);
  _Float16* rq_fij = halloc(64 * 512);
  _Float16* rq_nd  = halloc(64 * 512);
  _Float16* w_el   = halloc(2 * 64);
  _Float16* w_np   = halloc(44 * 64);
  _Float16* w_l1   = halloc(256 * 64);
  _Float16* w_l2   = halloc(1024 * 64);
  _Float16* w_l2a  = halloc(512 * 64);
  _Float16* w_l3   = halloc(512 * 64);
  _Float16* w_r1   = halloc(64 * 64);
  _Float16* w_r2   = halloc(64 * 64);
  _Float16* w_r3   = halloc(64 * 3);

  auto cvt = [&](const float* W, _Float16* Wt, int K, int N) {
    cvt_wT<<<g1((long)K * N), 256, 0, stream>>>(W, Wt, K, N);
  };
  cvt(rconv1.W_ni, c1_ni, 64, 512);   cvt(rconv1.W_nj, c1_nj, 64, 512);
  cvt(rconv1.W_fij, c1_fij, 64, 512); cvt(rconv1.W_node, c1_nd, 64, 512);
  cvt(rconv2.W_ni, c2_ni, 512, 256);  cvt(rconv2.W_nj, c2_nj, 512, 256);
  cvt(rconv2.W_fij, c2_fij, 512, 256);cvt(rconv2.W_node, c2_nd, 512, 1024);
  cvt(rrq.W_ni, rq_ni, 64, 512);      cvt(rrq.W_nj, rq_nj, 64, 512);
  cvt(rrq.W_fij, rq_fij, 64, 512);    cvt(rrq.W_node, rq_nd, 64, 512);
  cvt(edge_lin_W, w_el, 2, 64);
  cvt(nproj_W, w_np, 44, 64);
  cvt(lin1_W, w_l1, 256, 64);
  cvt(lin2_W, w_l2, 1024, 64);
  cvt(lin2a_W, w_l2a, 512, 64);
  cvt(lin3_W, w_l3, 512, 64);
  cvt(reg1_W, w_r1, 64, 64);
  cvt(reg2_W, w_r2, 64, 64);
  cvt(reg3_W, w_r3, 64, 3);

  EgatW conv1 = {c1_fij, c1_ni, c1_nj, c1_nd, rconv1.attn, rconv1.bias};
  EgatW conv2 = {c2_fij, c2_ni, c2_nj, c2_nd, rconv2.attn, rconv2.bias};
  EgatW rq    = {rq_fij, rq_ni, rq_nj, rq_nd, rrq.attn, rrq.bias};

  auto gemm = [&](const float* A, const _Float16* Wt, const float* b, float* C,
                  int M, int K, int Ncols, int act) {
    dim3 grid((M + 31) / 32, (Ncols + 127) / 128);
    gemm_wmma<<<grid, dim3(256), 0, stream>>>(A, Wt, b, C, M, K, Ncols, act);
  };

  auto egat = [&](const float* nf, const float* ef, const int* src, const int* dst,
                  const EgatW& P, int in_n, int in_e, int out_n, int out_e, int E,
                  float* fout, float* hout) {
    int D = H_ * out_e, HD = H_ * out_n;
    gemm(nf, P.W_ni, nullptr, fni, Nn, in_n, D, 0);
    gemm(nf, P.W_nj, nullptr, fnj, Nn, in_n, D, 0);
    gemm(ef, P.W_fij, nullptr, fout, E, in_e, D, 0);
    edge_combine<<<g1((long)E * D), 256, 0, stream>>>(fout, fni, fnj, P.bias, src, dst, E, D);
    attn_logits<<<g1((long)E * H_), 256, 0, stream>>>(fout, P.attn, eatt, E, out_e);
    fill_f32<<<g1((long)Nn * H_), 256, 0, stream>>>(emax, (long)Nn * H_, -INFINITY);
    seg_max<<<g1((long)E * H_), 256, 0, stream>>>(eatt, dst, emax, E);
    fix_max<<<g1((long)Nn * H_), 256, 0, stream>>>(emax, Nn * H_);
    fill_f32<<<g1((long)Nn * H_), 256, 0, stream>>>(esum, (long)Nn * H_, 0.f);
    exp_sum<<<g1((long)E * H_), 256, 0, stream>>>(eatt, emax, dst, exb, esum, E);
    attn_norm<<<g1((long)E * H_), 256, 0, stream>>>(exb, esum, dst, E);
    gemm(nf, P.W_node, nullptr, hsrc, Nn, in_n, HD, 0);
    fill_f32<<<g1((long)Nn * HD), 256, 0, stream>>>(hout, (long)Nn * HD, 0.f);
    scatter_h<<<g1((long)E * HD), 256, 0, stream>>>(hsrc, exb, src, dst, hout, E, HD, out_n);
  };

  // ---------------- forward pass ----------------
  gemm(nodeDat, w_np, nproj_b, node, Nn, IN_NODE, OUT_, 0);
  gemm(edgeDat, w_el, edge_lin_b, edge, E1_, IN_EDGE, OUT_, 0);
  gemm(edgeDat + (long)E1_ * IN_EDGE, w_el, edge_lin_b, elink, E2_, IN_EDGE, OUT_, 0);

  for (int it = 0; it < 2; ++it) {
    // conv1: in 64/64 -> out_n 128, out_e 128
    egat(node, edge, src1, dst1, conv1, OUT_, OUT_, 2 * OUT_, 2 * OUT_, E1_, ftmp, hA);
    relu_ip<<<g1((long)Nn * 512), 256, 0, stream>>>(hA, (long)Nn * 512);
    relu_ip<<<g1((long)E1_ * 512), 256, 0, stream>>>(ftmp, (long)E1_ * 512);
    // conv2: in 512/512 -> out_n 256, out_e 64
    egat(hA, ftmp, src1, dst1, conv2, 8 * OUT_, 8 * OUT_, 4 * OUT_, OUT_, E1_, fout2, hB);
    gemm(fout2, w_l1, lin1_b, edge, E1_, 4 * OUT_, OUT_, 1);
    gemm(hB, w_l2, lin2_b, node, Nn, 16 * OUT_, OUT_, 1);
    // rq: in 64/64 -> out_n 128, out_e 128 over graph 2
    egat(node, elink, src2, dst2, rq, OUT_, OUT_, 2 * OUT_, 2 * OUT_, E2_, ftmp, hA);
    gemm(ftmp, w_l3, lin3_b, elink, E2_, 8 * OUT_, OUT_, 1);
    gemm(hA, w_l2a, lin2a_b, nodet, Nn, 8 * OUT_, OUT_, 1);
    add_ip<<<g1((long)Nn * OUT_), 256, 0, stream>>>(node, nodet, (long)Nn * OUT_);
  }

  // regression head
  col_max<<<dim3(1), dim3(64), 0, stream>>>(node + (long)NR_ * OUT_, feat, Nn - NR_, OUT_);
  gemm(feat, w_r1, reg1_b, x1, 1, OUT_, OUT_, 1);
  gemm(x1, w_r2, reg2_b, x2, 1, OUT_, OUT_, 1);
  gemm(x2, w_r3, reg3_b, loc, 1, OUT_, 3, 0);

  // outputs: edge (E1 x 64), node[NR:] (1500 x 64), loc (3)
  float* out = (float*)d_out;
  copy_f32<<<g1((long)E1_ * OUT_), 256, 0, stream>>>(edge, out, (long)E1_ * OUT_);
  copy_f32<<<g1((long)(Nn - NR_) * OUT_), 256, 0, stream>>>(
      node + (long)NR_ * OUT_, out + (long)E1_ * OUT_, (long)(Nn - NR_) * OUT_);
  copy_f32<<<g1(3), 256, 0, stream>>>(loc, out + (long)E1_ * OUT_ + (long)(Nn - NR_) * OUT_, 3);
}
